// MultiHeadSelfAttention_25331717112082
// MI455X (gfx1250) — matrix operand order repaired and checked by reading, not compiled, not run on hardware
//
#include <hip/hip_runtime.h>

// ---------------------------------------------------------------------------
// Multi-head self attention (B=4, T=2048, C=1024, H=16, Dh=64), fp32 I/O.
// bf16 WMMA (v_wmma_f32_16x16x32_bf16) for all matmul stages, fp32 accum,
// fp32 online softmax. GEMMs use TDM (tensor_load_to_lds) double-buffered
// LDS staging; attention is a flash-attention single pass.
// ---------------------------------------------------------------------------

typedef __bf16    v16bf __attribute__((ext_vector_type(16)));
typedef float     v8f   __attribute__((ext_vector_type(8)));
typedef unsigned  v4u   __attribute__((ext_vector_type(4)));
typedef int       v8i   __attribute__((ext_vector_type(8)));
typedef int       v4i   __attribute__((ext_vector_type(4)));

#define BSZ   4
#define T_SEQ 2048
#define CDIM  1024
#define NH    16
#define DH    64
#define M_TOT (BSZ * T_SEQ)   // 8192
#define LDSPITCH 40           // bf16 elements per LDS row (64B data + 16B pad)

static __device__ __forceinline__ v8f wmma_bf16(v16bf a, v16bf b, v8f c) {
  return __builtin_amdgcn_wmma_f32_16x16x32_bf16(
      false, a, false, b, (short)0, c, false, false);
}

static __device__ __forceinline__ v16bf ld16bf(const __bf16* p0, const __bf16* p1) {
  union { v16bf bf; uint4 q[2]; } u;
  u.q[0] = *reinterpret_cast<const uint4*>(p0);   // elements 0..7  -> VGPR 0..3
  u.q[1] = *reinterpret_cast<const uint4*>(p1);   // elements 8..15 -> VGPR 4..7
  return u.bf;
}

// A-tile 16x32 (MxK) from row-major [lda] at (m0,k0). ISA 7.12.2 layout:
// lanes 0-15 row=lane, K 0..7 / 16..23; lanes 16-31 K 8..15 / 24..31.
static __device__ __forceinline__ v16bf ldA(const __bf16* A, int lda, int m0, int k0,
                                            int r, int h) {
  const __bf16* p0 = A + (size_t)(m0 + r) * lda + k0 + h * 8;
  return ld16bf(p0, p0 + 16);
}
// B-tile 32x16 (KxN) from row-major S[n][k] (B^T in memory) at (n0,k0):
// lanes 0-15 col=lane, K 0..7 / 16..23; lanes 16-31 K 8..15 / 24..31.
static __device__ __forceinline__ v16bf ldB(const __bf16* S, int ldS, int n0, int k0,
                                            int c, int h) {
  const __bf16* p0 = S + (size_t)(n0 + c) * ldS + k0 + h * 8;
  return ld16bf(p0, p0 + 16);
}

static __device__ __forceinline__ float rowmax16(float v) {
#pragma unroll
  for (int off = 1; off < 16; off <<= 1) v = fmaxf(v, __shfl_xor(v, off, 32));
  return v;
}
static __device__ __forceinline__ float rowsum16(float v) {
#pragma unroll
  for (int off = 1; off < 16; off <<= 1) v += __shfl_xor(v, off, 32);
  return v;
}

// ---------------------------------------------------------------------------
// Tensor Data Mover: stage a 2D tile (128 rows x 32 bf16) from a row-major
// global panel (row stride 1024 elements) into LDS at `lds_off`, with TDM
// padding 4 DWORDs every 16 DWORDs so each LDS row lands at a 40-element
// pitch (bank-conflict-free ds_load_b128 fragment reads).
// ---------------------------------------------------------------------------
static __device__ __forceinline__ void tdm_load_panel(unsigned lds_off,
                                                      const __bf16* gptr) {
  union { v4u v; unsigned w[4]; } g0;
  const unsigned long long ga = (unsigned long long)(uintptr_t)gptr;
  g0.w[0] = 1u;                                   // count=1 (valid), user mode
  g0.w[1] = lds_off;                              // lds_addr (bytes)
  g0.w[2] = (unsigned)(ga & 0xffffffffu);         // global_addr[31:0]
  g0.w[3] = (unsigned)((ga >> 32) & 0x01ffffffu)  // global_addr[56:32]
            | (2u << 30);                         // type = 2 ("image")
  union { v8i v; unsigned w[8]; } g1;
  const unsigned td0 = 1u << 20, td1 = 1u << 20;  // tensor dims (no OOB hit)
  g1.w[0] = (1u << 16)          // data_size = 2 bytes
          | (1u << 20)          // pad_enable
          | (3u << 22)          // pad_interval: every 16 DWORDs (64B row)
          | (3u << 25);         // pad_amount: 4 DWORDs (16B) -> 80B pitch
  g1.w[1] = (td0 & 0xffffu) << 16;                      // tensor_dim0 lo16
  g1.w[2] = (td0 >> 16) | ((td1 & 0xffffu) << 16);      // dim0 hi / dim1 lo
  g1.w[3] = (td1 >> 16) | (32u << 16);                  // dim1 hi / tile_dim0=32
  g1.w[4] = 128u;                                       // tile_dim1=128 rows
  g1.w[5] = (unsigned)CDIM;                             // dim0_stride lo32
  g1.w[6] = 0u;
  g1.w[7] = 0u;
  const v4i z4 = {0, 0, 0, 0};
  const v8i z8 = {0, 0, 0, 0, 0, 0, 0, 0};
  __builtin_amdgcn_tensor_load_to_lds(g0.v, g1.v, z4, z4, z8, 0);
}

static __device__ __forceinline__ unsigned lds_offset_of(const void* p) {
  return (unsigned)(uintptr_t)p;  // LDS aperture: low 32 bits = LDS byte offset
}

// ---------------------------------------------------------------------------
// fp32 -> bf16 convert / tiled transpose
// ---------------------------------------------------------------------------
__global__ void __launch_bounds__(256) k_cvt(const float* __restrict__ s,
                                             __bf16* __restrict__ d, int n) {
  int i = blockIdx.x * 256 + threadIdx.x;
  if (i < n) d[i] = (__bf16)s[i];
}

// s: [1024][N] fp32 row-major, d: [N][1024] bf16. Grid (N/32, 1024/32), 256 thr.
__global__ void __launch_bounds__(256) k_transpose(const float* __restrict__ s,
                                                   __bf16* __restrict__ d, int N) {
  __shared__ float tile[32][33];
  const int n0 = blockIdx.x * 32, k0 = blockIdx.y * 32;
  const int tx = threadIdx.x & 31, ty = threadIdx.x >> 5;  // 32 x 8
#pragma unroll
  for (int i = 0; i < 4; ++i)
    tile[ty + i * 8][tx] = s[(size_t)(k0 + ty + i * 8) * N + n0 + tx];
  __syncthreads();
#pragma unroll
  for (int i = 0; i < 4; ++i)
    d[(size_t)(n0 + ty + i * 8) * CDIM + k0 + tx] = (__bf16)tile[tx][ty + i * 8];
}

// ---------------------------------------------------------------------------
// WMMA GEMM core: block = 256 threads (8 waves: 4M x 2N), block tile 128x128,
// K chunks of 32 staged by TDM into double-buffered LDS panels.
// ---------------------------------------------------------------------------
struct GemmAcc { v8f a[2][4]; };

static __device__ __forceinline__ GemmAcc gemm_core(const __bf16* __restrict__ A,
                                                    const __bf16* __restrict__ BT,
                                                    __bf16 (*Abuf)[128][LDSPITCH],
                                                    __bf16 (*Bbuf)[128][LDSPITCH]) {
  const int lane = threadIdx.x & 31;
  const int wave = threadIdx.x >> 5;
  const int wm = wave & 3, wn = wave >> 2;
  const int r = lane & 15, h = lane >> 4;

  const __bf16* Ag = A + (size_t)(blockIdx.x * 128) * CDIM;   // panel base
  const __bf16* Bg = BT + (size_t)(blockIdx.y * 128) * CDIM;

  GemmAcc acc;
  const v8f vzero = {0.f, 0.f, 0.f, 0.f, 0.f, 0.f, 0.f, 0.f};
#pragma unroll
  for (int i = 0; i < 2; ++i)
#pragma unroll
    for (int j = 0; j < 4; ++j) acc.a[i][j] = vzero;

  // Prologue: stage chunk 0 into buffer 0.
  if (wave == 0) {
    tdm_load_panel(lds_offset_of(&Abuf[0][0][0]), Ag);
    tdm_load_panel(lds_offset_of(&Bbuf[0][0][0]), Bg);
  }
  __builtin_amdgcn_s_wait_tensorcnt(0);
  __syncthreads();

  const int nchunk = CDIM / 32;
  for (int i = 0; i < nchunk; ++i) {
    const int cur = i & 1;
    if (wave == 0 && (i + 1) < nchunk) {
      tdm_load_panel(lds_offset_of(&Abuf[cur ^ 1][0][0]), Ag + (i + 1) * 32);
      tdm_load_panel(lds_offset_of(&Bbuf[cur ^ 1][0][0]), Bg + (i + 1) * 32);
    }
    // Fragments from LDS (ds_load_b128), then WMMA.
    const __bf16* a0p = &Abuf[cur][wm * 32 + r][h * 8];
    const __bf16* a1p = &Abuf[cur][wm * 32 + 16 + r][h * 8];
    v16bf a0 = ld16bf(a0p, a0p + 16);
    v16bf a1 = ld16bf(a1p, a1p + 16);
#pragma unroll
    for (int j = 0; j < 4; ++j) {
      const __bf16* bp = &Bbuf[cur][wn * 64 + j * 16 + r][h * 8];
      v16bf b = ld16bf(bp, bp + 16);
      acc.a[0][j] = wmma_bf16(a0, b, acc.a[0][j]);
      acc.a[1][j] = wmma_bf16(a1, b, acc.a[1][j]);
    }
    __builtin_amdgcn_s_wait_tensorcnt(0);  // next chunk resident (wave0)
    __syncthreads();                       // all reads of cur done before reuse
  }
  return acc;
}

// ---------------------------------------------------------------------------
// QKV projection: [8192,1024] x [1024,3072] (+bias), scatter to
//   Qb [bh][t][d], Kb [bh][t][d], VTb [bh][d][t]  (bf16)
// ---------------------------------------------------------------------------
__global__ void __launch_bounds__(256)
k_gemm_qkv(const __bf16* __restrict__ A, const __bf16* __restrict__ BT,
           const float* __restrict__ bias, __bf16* __restrict__ Qb,
           __bf16* __restrict__ Kb, __bf16* __restrict__ VTb) {
  __shared__ __align__(16) __bf16 Abuf[2][128][LDSPITCH];
  __shared__ __align__(16) __bf16 Bbuf[2][128][LDSPITCH];

  GemmAcc acc = gemm_core(A, BT, Abuf, Bbuf);

  const int lane = threadIdx.x & 31;
  const int wave = threadIdx.x >> 5;
  const int wm = wave & 3, wn = wave >> 2;
  const int m0 = blockIdx.x * 128 + wm * 32;
  const int n0 = blockIdx.y * 128 + wn * 64;
  const int r = lane & 15, h = lane >> 4;

#pragma unroll
  for (int i = 0; i < 2; ++i) {
#pragma unroll
    for (int j = 0; j < 4; ++j) {
      const int col = n0 + j * 16 + r;       // 0..3071
      const float bv = bias[col];
      const int sect = col >> 10;            // 0:Q 1:K 2:V (uniform per tile)
      const int cc = col & 1023;
      const int hh = cc >> 6;                // head
      const int d = cc & 63;                 // feature in head
#pragma unroll
      for (int e = 0; e < 8; ++e) {
        const int m = m0 + i * 16 + e + 8 * h;
        const int b_ = m >> 11, t = m & 2047;
        const __bf16 v = (__bf16)(acc.a[i][j][e] + bv);
        const size_t bh = (size_t)(b_ * NH + hh);
        if (sect == 0)      Qb[(bh * T_SEQ + t) * DH + d] = v;
        else if (sect == 1) Kb[(bh * T_SEQ + t) * DH + d] = v;
        else                VTb[(bh * DH + d) * T_SEQ + t] = v;  // V transposed
      }
    }
  }
}

// ---------------------------------------------------------------------------
// Output projection: AO[8192,1024] x Wout[1024,1024] (+bias) -> fp32 out
// ---------------------------------------------------------------------------
__global__ void __launch_bounds__(256)
k_gemm_out(const __bf16* __restrict__ A, const __bf16* __restrict__ BT,
           const float* __restrict__ bias, float* __restrict__ out) {
  __shared__ __align__(16) __bf16 Abuf[2][128][LDSPITCH];
  __shared__ __align__(16) __bf16 Bbuf[2][128][LDSPITCH];

  GemmAcc acc = gemm_core(A, BT, Abuf, Bbuf);

  const int lane = threadIdx.x & 31;
  const int wave = threadIdx.x >> 5;
  const int wm = wave & 3, wn = wave >> 2;
  const int m0 = blockIdx.x * 128 + wm * 32;
  const int n0 = blockIdx.y * 128 + wn * 64;
  const int r = lane & 15, h = lane >> 4;

#pragma unroll
  for (int i = 0; i < 2; ++i) {
#pragma unroll
    for (int j = 0; j < 4; ++j) {
      const int col = n0 + j * 16 + r;
      const float bv = bias[col];
#pragma unroll
      for (int e = 0; e < 8; ++e) {
        const int m = m0 + i * 16 + e + 8 * h;
        out[(size_t)m * CDIM + col] = acc.a[i][j][e] + bv;
      }
    }
  }
}

// ---------------------------------------------------------------------------
// Flash attention. Grid: (bh=64, qblock=32). Block = 128 threads = 4 waves.
// Wave w handles 16 queries [qb*64 + w*16, +16). Keys in tiles of 32.
// ---------------------------------------------------------------------------
__global__ void __launch_bounds__(128)
k_attn(const __bf16* __restrict__ Qb, const __bf16* __restrict__ Kb,
       const __bf16* __restrict__ VTb, __bf16* __restrict__ AO) {
  __shared__ __align__(16) __bf16 plds[4][16][40];  // per-wave P tile, padded

  const int lane = threadIdx.x & 31;
  const int w = threadIdx.x >> 5;
  const int bh = blockIdx.x;
  const int b_ = bh >> 4, hh = bh & 15;
  const int qb = blockIdx.y;
  const int q0 = qb * 64 + w * 16;
  const int r = lane & 15, h = lane >> 4;

  const __bf16* Qbase = Qb + (size_t)bh * T_SEQ * DH;
  const __bf16* Kbase = Kb + (size_t)bh * T_SEQ * DH;
  const __bf16* Vbase = VTb + (size_t)bh * DH * T_SEQ;

  const v16bf qa0 = ldA(Qbase, DH, q0, 0, r, h);
  const v16bf qa1 = ldA(Qbase, DH, q0, 32, r, h);

  const v8f vzero = {0.f, 0.f, 0.f, 0.f, 0.f, 0.f, 0.f, 0.f};
  v8f o[4];
#pragma unroll
  for (int t = 0; t < 4; ++t) o[t] = vzero;
  float mrow[8], lrow[8];
#pragma unroll
  for (int e = 0; e < 8; ++e) { mrow[e] = -1.0e30f; lrow[e] = 0.0f; }

  const float scale = 0.125f;  // 1/sqrt(64)
  const int kend = qb * 64 + 64;

  for (int kt = 0; kt < kend; kt += 32) {
    // S = Q K^T : two 16x16 tiles (keys kt..kt+31)
    v8f s[2];
#pragma unroll
    for (int j = 0; j < 2; ++j) {
      v16bf kb0 = ldB(Kbase, DH, kt + j * 16, 0, r, h);
      v16bf kb1 = ldB(Kbase, DH, kt + j * 16, 32, r, h);
      v8f t = wmma_bf16(qa0, kb0, vzero);
      s[j] = wmma_bf16(qa1, kb1, t);
    }
    // scale + causal mask
#pragma unroll
    for (int j = 0; j < 2; ++j) {
#pragma unroll
      for (int e = 0; e < 8; ++e) {
        const int kg = kt + j * 16 + r;
        const int qg = q0 + e + 8 * h;
        const float sv = s[j][e] * scale;
        s[j][e] = (kg <= qg) ? sv : -1.0e30f;
      }
    }
    // online softmax (row reductions across the 16-lane half)
#pragma unroll
    for (int e = 0; e < 8; ++e) {
      float mx = rowmax16(fmaxf(s[0][e], s[1][e]));
      const float mn = fmaxf(mrow[e], mx);
      const float sc = __expf(mrow[e] - mn);
      const float p0 = __expf(s[0][e] - mn);
      const float p1 = __expf(s[1][e] - mn);
      lrow[e] = lrow[e] * sc + rowsum16(p0 + p1);
      mrow[e] = mn;
#pragma unroll
      for (int t = 0; t < 4; ++t) o[t][e] *= sc;
      s[0][e] = p0;
      s[1][e] = p1;
    }
    // P (C-layout f32) -> LDS bf16 -> A-layout fragment
#pragma unroll
    for (int e = 0; e < 8; ++e) {
      plds[w][e + 8 * h][r] = (__bf16)s[0][e];
      plds[w][e + 8 * h][16 + r] = (__bf16)s[1][e];
    }
    asm volatile("s_wait_dscnt 0" ::: "memory");
    const __bf16* pr = &plds[w][r][0];
    const v16bf pa = ld16bf(pr + h * 8, pr + 16 + h * 8);
    // O += P V  (V^T rows are features: contiguous along keys)
#pragma unroll
    for (int nf = 0; nf < 4; ++nf) {
      v16bf vb = ldB(Vbase, T_SEQ, nf * 16, kt, r, h);
      o[nf] = wmma_bf16(pa, vb, o[nf]);
    }
  }

  // normalize and store AO [8192][1024] bf16
#pragma unroll
  for (int e = 0; e < 8; ++e) {
    const float inv = 1.0f / lrow[e];
    const int tq = q0 + e + 8 * h;
    const size_t row = (size_t)(b_ * T_SEQ + tq) * CDIM + hh * DH;
#pragma unroll
    for (int nf = 0; nf < 4; ++nf)
      AO[row + nf * 16 + r] = (__bf16)(o[nf][e] * inv);
  }
}

// ---------------------------------------------------------------------------
// Host launcher
// ---------------------------------------------------------------------------
extern "C" void kernel_launch(void* const* d_in, const int* in_sizes, int n_in,
                              void* d_out, int out_size, void* d_ws, size_t ws_size,
                              hipStream_t stream) {
  const float* x     = (const float*)d_in[0];  // [4,2048,1024]
  const float* w_qkv = (const float*)d_in[1];  // [1024,3072]
  const float* b_qkv = (const float*)d_in[2];  // [3072]
  const float* w_out = (const float*)d_in[3];  // [1024,1024]
  const float* b_out = (const float*)d_in[4];  // [1024]
  float* out = (float*)d_out;                  // [8192,1024]

  char* ws = (char*)d_ws;
  size_t off = 0;
  __bf16* xb    = (__bf16*)(ws + off); off += (size_t)M_TOT * CDIM * 2;
  __bf16* wqkvT = (__bf16*)(ws + off); off += (size_t)3 * CDIM * CDIM * 2;
  __bf16* woutT = (__bf16*)(ws + off); off += (size_t)CDIM * CDIM * 2;
  __bf16* Qb    = (__bf16*)(ws + off); off += (size_t)M_TOT * CDIM * 2;
  __bf16* Kb    = (__bf16*)(ws + off); off += (size_t)M_TOT * CDIM * 2;
  __bf16* VTb   = (__bf16*)(ws + off); off += (size_t)M_TOT * CDIM * 2;
  __bf16* AO    = (__bf16*)(ws + off); off += (size_t)M_TOT * CDIM * 2;

  // 1) precision conversions
  k_cvt<<<(M_TOT * CDIM + 255) / 256, 256, 0, stream>>>(x, xb, M_TOT * CDIM);
  k_transpose<<<dim3((3 * CDIM) / 32, CDIM / 32), 256, 0, stream>>>(w_qkv, wqkvT,
                                                                    3 * CDIM);
  k_transpose<<<dim3(CDIM / 32, CDIM / 32), 256, 0, stream>>>(w_out, woutT, CDIM);

  // 2) QKV projection (TDM-staged WMMA GEMM)
  k_gemm_qkv<<<dim3(M_TOT / 128, (3 * CDIM) / 128), 256, 0, stream>>>(
      xb, wqkvT, b_qkv, Qb, Kb, VTb);

  // 3) flash attention
  k_attn<<<dim3(BSZ * NH, T_SEQ / 64), 128, 0, stream>>>(Qb, Kb, VTb, AO);

  // 4) output projection (TDM-staged WMMA GEMM)
  k_gemm_out<<<dim3(M_TOT / 128, CDIM / 128), 256, 0, stream>>>(AO, woutT, b_out,
                                                                out);
}
